// GaussianVectorQuantizer_9268539424779
// MI455X (gfx1250) — compile-verified, matrix operand order checked by reading, not run on hardware
//
#include <hip/hip_runtime.h>
#include <hip/hip_bf16.h>

// ---------------------------------------------------------------------------
// GaussianVectorQuantizer forward, fused for MI455X (gfx1250, wave32, WMMA,
// Tensor Data Mover).
// ---------------------------------------------------------------------------

typedef __attribute__((ext_vector_type(16))) __bf16 v16bf;
typedef __attribute__((ext_vector_type(8)))  float  v8f;
typedef unsigned int u32x4 __attribute__((ext_vector_type(4)));
typedef int          i32x8 __attribute__((ext_vector_type(8)));
typedef int          i32x4 __attribute__((ext_vector_type(4)));

union BFFrag {                 // one WMMA bf16 A/B operand (16 halves / lane)
    uint4           q[2];      // two 16B chunks
    unsigned short  s[16];
    v16bf           v;
};

// Problem constants (fixed by the reference setup_inputs()).
#define DZ        256          // dim_z
#define KCODES    1024         // codebook size
#define BSZ       64           // batch
#define NTOK      65536        // bs*W*H
#define TILE      32           // tokens per workgroup
#define NWG       (NTOK / TILE)

// d_out flat layout: loss | z_to_decoder | perplexity | avg_probs | indices
#define Z_OFF     1
#define PERP_OFF  (1 + 16777216)
#define AVG_OFF   (PERP_OFF + 1)
#define IDX_OFF   (AVG_OFF + KCODES)

// LDS layout (bytes). sZt (f32 TDM staging tile) aliases sS: sZt is consumed
// in phase 1 before the logits are produced in phase 2.
#define SS_STRIDE 1032                         // f32 logits row stride
#define SE_STRIDE 1040                         // bf16 encodings row stride
#define SA_STRIDE 264                          // bf16 z-tile row stride
#define OFF_SS    0
#define OFF_SE    (OFF_SS + TILE*SS_STRIDE*4)          // 132096
#define OFF_SA    (OFF_SE + TILE*SE_STRIDE*2)          // +66560
#define OFF_AVG   (OFF_SA + TILE*SA_STRIDE*2)          // +16896
#define OFF_ZI    (OFF_AVG + KCODES*4)                 // +4096
#define OFF_RED   (OFF_ZI + TILE*4)                    // +128
#define SMEM_BYTES (OFF_RED + 64)                      // ~215 KB < 320 KB

__device__ __forceinline__ unsigned short f2bf(float x) {
    union { float f; unsigned u; } c; c.f = x;
    unsigned u = c.u;
    unsigned r = (u + 0x7fffu + ((u >> 16) & 1u)) >> 16;  // RNE
    return (unsigned short)r;
}

// Low 32 bits of an LDS-aperture flat address are the LDS byte offset.
__device__ __forceinline__ unsigned lds_off(const void* p) {
    return (unsigned)(unsigned long long)(uintptr_t)p;
}

// ---------------------------------------------------------------------------
// Prep: codebook f32 -> bf16 (row-major + transposed), norms, zero accums.
// ---------------------------------------------------------------------------
__global__ __launch_bounds__(64)
void gvq_prep(const float* __restrict__ cbf32,
              unsigned short* __restrict__ cb,     // [K][DZ]
              unsigned short* __restrict__ cbT,    // [DZ][K]
              float* __restrict__ cnorm,
              float* __restrict__ avgpAcc,
              float* __restrict__ scal) {
    __shared__ float red[64];
    const int k = blockIdx.x;
    if (k == 0 && threadIdx.x < 2) scal[threadIdx.x] = 0.0f;
    const float* row = cbf32 + (size_t)k * DZ;
    unsigned short* orow = cb + (size_t)k * DZ;
    float s = 0.0f;
    for (int i = threadIdx.x; i < DZ; i += 64) {
        float v = row[i];
        s += v * v;
        unsigned short h = f2bf(v);
        orow[i] = h;
        cbT[(size_t)i * KCODES + k] = h;
    }
    red[threadIdx.x] = s;
    __syncthreads();
    for (int st = 32; st > 0; st >>= 1) {
        if (threadIdx.x < st) red[threadIdx.x] += red[threadIdx.x + st];
        __syncthreads();
    }
    if (threadIdx.x == 0) { cnorm[k] = red[0]; avgpAcc[k] = 0.0f; }
}

// ---------------------------------------------------------------------------
// Main fused kernel: one WG = 32 tokens, 8 waves of 32 lanes.
// ---------------------------------------------------------------------------
__global__ __launch_bounds__(256)
void gvq_main(const float* __restrict__ z,
              const float* __restrict__ varq,
              const float* __restrict__ gu,
              const unsigned short* __restrict__ cb,
              const unsigned short* __restrict__ cbT,
              const float* __restrict__ cnorm,
              float* __restrict__ avgpAcc,
              float* __restrict__ scal,
              float* __restrict__ out) {
    extern __shared__ char smem[];
    float*          sS   = (float*)(smem + OFF_SS);            // logits [32][1032]
    float*          sZt  = (float*)(smem + OFF_SS);            // f32 z tile [256][32] (alias)
    unsigned short* sE   = (unsigned short*)(smem + OFF_SE);   // enc bf16 [32][1040]
    unsigned short* sA   = (unsigned short*)(smem + OFF_SA);   // z bf16 [32][264]
    float*          sAvg = (float*)(smem + OFF_AVG);           // [1024]
    float*          sZi  = (float*)(smem + OFF_ZI);            // [32]
    float*          sRed = (float*)(smem + OFF_RED);           // [8]

    const int lane = threadIdx.x & 31;
    const int wave = threadIdx.x >> 5;
    const int tile = blockIdx.x;
    const int t0   = tile * TILE;
    const int b    = t0 >> 10;          // 1024 tokens per batch element
    const int wh0  = t0 & 1023;
    const float p  = 1.0f / fmaxf(varq[0], 1e-10f);

    // ---- phase 0: zero LDS avg-prob accumulator ----
    for (int e = threadIdx.x; e < KCODES; e += 256) sAvg[e] = 0.0f;

    // ---- phase 1a: Tensor Data Mover pulls the 2-D z tile into LDS ----
    // Tile: 32 consecutive tokens (contiguous f32) x 256 feature lines with a
    // 1024-element line stride -> sZt[256][32] f32, one DMA descriptor.
    const float* zb = z + (size_t)b * DZ * 1024 + wh0;
    if (wave == 0) {
        unsigned long long ga = (unsigned long long)(uintptr_t)zb;
        u32x4 g0;
        g0[0] = 1u;                                   // count=1, user descriptor
        g0[1] = lds_off(sZt);                         // lds_addr
        g0[2] = (unsigned)(ga & 0xffffffffu);         // global_addr[95:64]
        g0[3] = (unsigned)((ga >> 32) & 0x01ffffffu)  // global_addr[120:96]
              | 0x80000000u;                          // type=2 ("image")
        i32x8 g1;
        g1[0] = 0x00020000;                           // data_size=2 (4B), no mask
        g1[1] = (32 & 0xffff) << 16;                  // tensor_dim0 = 32
        g1[2] = (256 & 0xffff) << 16;                 // tensor_dim1 = 256
        g1[3] = (32 & 0xffff) << 16;                  // tile_dim0 = 32
        g1[4] = 256;                                  // tile_dim1 = 256
        g1[5] = 1024;                                 // tensor_dim0_stride = 1024
        g1[6] = 0;
        g1[7] = 0;
        i32x4 g2 = {0, 0, 0, 0};                      // 2-D tensor: groups 2/3 unused
        i32x4 g3 = {0, 0, 0, 0};
        i32x8 g4 = {0, 0, 0, 0, 0, 0, 0, 0};          // unused trailing group (clang-23 form)
        __builtin_amdgcn_tensor_load_to_lds(g0, g1, g2, g3, g4, 0);
        __builtin_amdgcn_s_wait_tensorcnt(0);
    }
    __syncthreads();

    // ---- phase 1b: LDS transpose + f32->bf16 into WMMA-A layout ----
    for (int e = threadIdx.x; e < TILE * DZ; e += 256) {
        int d = e >> 5, r = e & 31;
        sA[r * SA_STRIDE + d] = f2bf(sZt[d * 32 + r]);
    }
    __syncthreads();

    // ---- phase 2: GEMM1 logits  l = p*(z.c) - 0.5*p*||c||^2 ----
    const int mr   = lane & 15;
    const int khlf = (lane < 16) ? 0 : 8;            // K split per lane-half (ISA 7.12.2)
    const int rb0  = (lane < 16) ? 0 : 8;            // C-tile row base per lane-half
    for (int ns = 0; ns < 8; ++ns) {
        const int n0  = wave * 128 + ns * 16;
        const int col = n0 + mr;
        v8f acc0 = {}, acc1 = {};
        const unsigned short* cbrow = cb + (size_t)col * DZ;
        __builtin_prefetch(cbrow, 0, 0);
        #pragma unroll
        for (int d0 = 0; d0 < DZ; d0 += 32) {
            BFFrag bf, a0, a1;
            bf.q[0] = *(const uint4*)(cbrow + d0 + khlf);
            bf.q[1] = *(const uint4*)(cbrow + d0 + khlf + 16);
            const unsigned short* a0p = sA + mr * SA_STRIDE + d0 + khlf;
            const unsigned short* a1p = sA + (16 + mr) * SA_STRIDE + d0 + khlf;
            a0.q[0] = *(const uint4*)a0p;  a0.q[1] = *(const uint4*)(a0p + 16);
            a1.q[0] = *(const uint4*)a1p;  a1.q[1] = *(const uint4*)(a1p + 16);
            acc0 = __builtin_amdgcn_wmma_f32_16x16x32_bf16(false, a0.v, false, bf.v,
                                                           (short)0, acc0, false, false);
            acc1 = __builtin_amdgcn_wmma_f32_16x16x32_bf16(false, a1.v, false, bf.v,
                                                           (short)0, acc1, false, false);
        }
        const float bias = 0.5f * p * cnorm[col];
        #pragma unroll
        for (int r = 0; r < 8; ++r) {
            sS[(r + rb0) * SS_STRIDE + col]      = p * acc0[r] - bias;
            sS[(16 + r + rb0) * SS_STRIDE + col] = p * acc1[r] - bias;
        }
    }
    __syncthreads();

    // ---- phase 3: per-row online softmax stats, gumbel encodings ----
    float kldd = 0.0f;
    for (int rr = 0; rr < 4; ++rr) {
        const int r = wave * 4 + rr;
        const size_t n = (size_t)t0 + r;
        const float* srow = sS + r * SS_STRIDE;
        // sweep 1 (LDS only): running max m, Z=sum e^{l-m}, T=sum e^{l-m}(l-m), argmax
        float mx = -3.0e38f, Z1 = 0.0f, T1 = 0.0f, bl = -3.0e38f;
        int bi = KCODES;
        for (int k = lane; k < KCODES; k += 32) {
            float l = srow[k];
            if (l > bl) { bl = l; bi = k; }
            if (l > mx) {
                float d = mx - l, s = __expf(d);
                T1 = s * (T1 + d * Z1);
                Z1 = s * Z1 + 1.0f;
                mx = l;
            } else {
                float w = __expf(l - mx);
                Z1 += w; T1 += w * (l - mx);
            }
        }
        for (int off = 16; off > 0; off >>= 1) {
            float mo  = __shfl_xor(mx, off, 32);
            float zo  = __shfl_xor(Z1, off, 32);
            float to  = __shfl_xor(T1, off, 32);
            float blo = __shfl_xor(bl, off, 32);
            int   bio = __shfl_xor(bi, off, 32);
            if (blo > bl || (blo == bl && bio < bi)) { bl = blo; bi = bio; }
            float M  = fmaxf(mx, mo);
            float s1 = __expf(mx - M), s2 = __expf(mo - M);
            T1 = s1 * (T1 + (mx - M) * Z1) + s2 * (to + (mo - M) * zo);
            Z1 = s1 * Z1 + s2 * zo;
            mx = M;
        }
        const float invZ1 = 1.0f / Z1;
        kldd += T1 * invZ1 - __logf(Z1);             // = sum_k p*log p for this row
        if (lane == 0) out[IDX_OFF + n] = (float)bi;
        // sweep 2: single gumbel read; unnormalized gumbel-softmax -> bf16 LDS
        const float* gurow = gu + n * (size_t)KCODES;
        __builtin_prefetch(gurow + lane, 0, 0);
        const float B2 = (mx + 23.03f) * 2.0f;       // safe upper bound of (l+g)/T
        float Z2 = 0.0f;
        unsigned short* erow = sE + r * SE_STRIDE;
        for (int k = lane; k < KCODES; k += 32) {
            float l = srow[k];
            float u = gurow[k];
            float g = -__logf(-__logf(u + 1e-10f) + 1e-10f);
            float e2 = __expf((l + g) * 2.0f - B2);  // 1/TEMPERATURE = 2
            Z2 += e2;
            erow[k] = f2bf(e2);
            atomicAdd(&sAvg[k], __expf(l - mx) * invZ1);
        }
        for (int off = 16; off > 0; off >>= 1) Z2 += __shfl_xor(Z2, off, 32);
        if (lane == 0) sZi[r] = 1.0f / Z2;           // folded into GEMM2 epilogue
    }
    if (lane == 0) sRed[wave] = kldd;
    __syncthreads();
    if (threadIdx.x == 0) {
        float s = 0.0f;
        for (int i = 0; i < 8; ++i) s += sRed[i];
        atomicAdd(&scal[0], s);
    }

    // ---- phase 4: GEMM2  z_q = (enc @ C) / Z2, fused NCHW store + KLD-cont ----
    // B fragments come straight from the transposed bf16 codebook (L2-resident):
    // two aligned 16B loads per fragment, no LDS staging, no barriers.
    v8f acc[2][2] = {{{}, {}}, {{}, {}}};
    const int d0col = wave * 32 + mr;
    for (int kc = 0; kc < KCODES; kc += 32) {
        if (kc + 32 < KCODES)
            __builtin_prefetch(cbT + (size_t)d0col * KCODES + kc + 32, 0, 0);
        #pragma unroll
        for (int j = 0; j < 2; ++j) {
            const unsigned short* btp =
                cbT + (size_t)(d0col + j * 16) * KCODES + kc + khlf;
            BFFrag bf;
            bf.q[0] = *(const uint4*)btp;  bf.q[1] = *(const uint4*)(btp + 16);
            #pragma unroll
            for (int m = 0; m < 2; ++m) {
                const unsigned short* ap = sE + (m * 16 + mr) * SE_STRIDE + kc + khlf;
                BFFrag a;
                a.q[0] = *(const uint4*)ap;  a.q[1] = *(const uint4*)(ap + 16);
                acc[m][j] = __builtin_amdgcn_wmma_f32_16x16x32_bf16(
                    false, a.v, false, bf.v, (short)0, acc[m][j], false, false);
            }
        }
    }
    float cont = 0.0f;
    #pragma unroll
    for (int m = 0; m < 2; ++m) {
        const int rb = m * 16 + rb0;                  // 8 consecutive token rows per lane
        float zi[8];
        #pragma unroll
        for (int r = 0; r < 8; ++r) zi[r] = sZi[rb + r];
        #pragma unroll
        for (int j = 0; j < 2; ++j) {
            const int d = d0col + j * 16;
            const size_t gbase = (size_t)b * DZ * 1024 + (size_t)d * 1024 + wh0 + rb;
            const float* zp = z + gbase;
            float* op = out + Z_OFF + gbase;
            #pragma unroll
            for (int r = 0; r < 8; ++r) {
                float v = acc[m][j][r] * zi[r];
                op[r] = v;
                float df = zp[r] - v;
                cont += df * df;
            }
        }
    }
    for (int off = 16; off > 0; off >>= 1) cont += __shfl_xor(cont, off, 32);
    __syncthreads();
    if (lane == 0) sRed[wave] = cont;
    __syncthreads();
    if (threadIdx.x == 0) {
        float s = 0.0f;
        for (int i = 0; i < 8; ++i) s += sRed[i];
        atomicAdd(&scal[1], s);
    }
    for (int e = threadIdx.x; e < KCODES; e += 256)
        atomicAdd(&avgpAcc[e], sAvg[e]);
}

// ---------------------------------------------------------------------------
// Finalize: avg_probs, perplexity, loss.
// ---------------------------------------------------------------------------
__global__ __launch_bounds__(256)
void gvq_finalize(const float* __restrict__ varq,
                  const float* __restrict__ avgpAcc,
                  const float* __restrict__ scal,
                  float* __restrict__ out) {
    __shared__ float red[256];
    float ent = 0.0f;
    for (int k = threadIdx.x; k < KCODES; k += 256) {
        float a = avgpAcc[k] * (1.0f / (float)NTOK);
        out[AVG_OFF + k] = a;
        ent += a * __logf(a + 1e-7f);
    }
    red[threadIdx.x] = ent;
    __syncthreads();
    for (int s = 128; s > 0; s >>= 1) {
        if (threadIdx.x < s) red[threadIdx.x] += red[threadIdx.x + s];
        __syncthreads();
    }
    if (threadIdx.x == 0) {
        float psum = 1.0f / fmaxf(varq[0], 1e-10f);
        out[0]        = scal[0] / (float)BSZ + 0.5f * psum * scal[1] / (float)BSZ;
        out[PERP_OFF] = __expf(-red[0]);
    }
}

// ---------------------------------------------------------------------------
extern "C" void kernel_launch(void* const* d_in, const int* in_sizes, int n_in,
                              void* d_out, int out_size, void* d_ws, size_t ws_size,
                              hipStream_t stream) {
    const float* z    = (const float*)d_in[0];
    const float* varq = (const float*)d_in[1];
    const float* cbf  = (const float*)d_in[2];
    const float* gu   = (const float*)d_in[3];
    float* out = (float*)d_out;

    char* ws = (char*)d_ws;
    unsigned short* cb  = (unsigned short*)ws;               // 512 KB row-major bf16
    unsigned short* cbT = (unsigned short*)(ws + 524288);    // 512 KB transposed bf16
    float* cnorm        = (float*)(ws + 1048576);            // 4 KB
    float* avgp         = (float*)(ws + 1052672);            // 4 KB
    float* scal         = (float*)(ws + 1056768);            // kldd, cont

    gvq_prep<<<KCODES, 64, 0, stream>>>(cbf, cb, cbT, cnorm, avgp, scal);
    gvq_main<<<NWG, 256, SMEM_BYTES, stream>>>(z, varq, gu, cb, cbT, cnorm, avgp, scal, out);
    gvq_finalize<<<1, 256, 0, stream>>>(varq, avgp, scal, out);
}
